// MergeLayer_48876727828694
// MI455X (gfx1250) — matrix operand order repaired and checked by reading, not compiled, not run on hardware
//
#include <hip/hip_runtime.h>
#include <hip/hip_bf16.h>
#include <hip/hip_fp16.h>

typedef __attribute__((ext_vector_type(16))) _Float16 v16h;
typedef __attribute__((ext_vector_type(8)))  float    v8f;

#define D        128
#define HIDDEN   128
#define N_EDGES_ 600000
#define NTILES   (HIDDEN / 16)    // 8 column tiles of the hidden layer
#define KSTEPS   ((2 * D) / 32)   // 8 K-steps of 32
#define WAVES_PB 8                // 256 threads per block
#define TILES    (N_EDGES_ / 16)  // 37500 edge tiles
#define PAIRS    (TILES / 2)      // 18750 tile-pairs (one per wave)

// ---------------------------------------------------------------------------
// Prep: repack W1 (f32, [256][128] row-major) into the exact WMMA B-fragment
// layout (f16), one v16h (32B) per lane per (k_step, n_tile):
//   B 32x16 f16 layout: lanes 0-15 hold K=0..7,16..23,
//   lanes 16-31 hold K=8..15,24..31; column N = lane & 15.
// Flat layout: Bp[s][nt][lane][16 halves]
// ---------------------------------------------------------------------------
__global__ void pack_w1_kernel(const float* __restrict__ W1,
                               _Float16* __restrict__ Bp) {
    int idx = blockIdx.x * blockDim.x + threadIdx.x;   // 8*8*32*16 = 32768
    if (idx >= KSTEPS * NTILES * 32 * 16) return;
    int e    = idx & 15;          // element within v16h
    int lane = (idx >> 4) & 31;
    int nt   = (idx >> 9) & 7;
    int s    = (idx >> 12) & 7;
    int klocal = ((lane < 16) ? 0 : 8) + (e < 8 ? e : e + 8); // K within 32-step
    int k = s * 32 + klocal;                            // 0..255
    int n = nt * 16 + (lane & 15);                      // 0..127
    Bp[idx] = (_Float16)W1[k * HIDDEN + n];
}

// Build an A fragment (16x32 f16) for one 16-edge tile from two 32B runs.
__device__ __forceinline__ v16h make_a_frag(const float* src) {
    float4 a0 = ((const float4*)src)[0];         // K = +0..3
    float4 a1 = ((const float4*)src)[1];         // K = +4..7
    float4 a2 = ((const float4*)(src + 16))[0];  // K = +16..19
    float4 a3 = ((const float4*)(src + 16))[1];  // K = +20..23
    v16h a;
    a[0]  = (_Float16)a0.x; a[1]  = (_Float16)a0.y;
    a[2]  = (_Float16)a0.z; a[3]  = (_Float16)a0.w;
    a[4]  = (_Float16)a1.x; a[5]  = (_Float16)a1.y;
    a[6]  = (_Float16)a1.z; a[7]  = (_Float16)a1.w;
    a[8]  = (_Float16)a2.x; a[9]  = (_Float16)a2.y;
    a[10] = (_Float16)a2.z; a[11] = (_Float16)a2.w;
    a[12] = (_Float16)a3.x; a[13] = (_Float16)a3.y;
    a[14] = (_Float16)a3.z; a[15] = (_Float16)a3.w;
    return a;
}

// ---------------------------------------------------------------------------
// Main: one wave = 2 tiles of 16 edges (M-blocking so each W1 B-fragment load
// feeds two WMMAs). 128x v_wmma_f32_16x16x32_f16 per wave, fused bias/ReLU,
// layer-2 matvec via shuffle butterfly, sigmoid, store.
// ---------------------------------------------------------------------------
__global__ __launch_bounds__(256) void edge_mlp_kernel(
    const float*    __restrict__ z,
    const int*      __restrict__ e0,
    const int*      __restrict__ e1,
    const _Float16* __restrict__ Bp,
    const float*    __restrict__ b1,
    const float*    __restrict__ W2,
    const float*    __restrict__ b2,
    float*          __restrict__ out) {

    const int lane = threadIdx.x & 31;
    const int wave = threadIdx.x >> 5;
    const int pair = blockIdx.x * WAVES_PB + wave;
    if (pair >= PAIRS) return;                 // whole-wave exit: EXEC stays all-1s

    const int tile0 = pair * 2;
    const int m     = lane & 15;               // A-fragment row M for this lane
    const int edgeA = tile0 * 16 + m;
    const int edgeB = edgeA + 16;

    const float* rowA0 = z + (long long)e0[edgeA] * D;  // x_i tile0
    const float* rowB0 = z + (long long)e1[edgeA] * D;  // x_j tile0
    const float* rowA1 = z + (long long)e0[edgeB] * D;  // x_i tile1
    const float* rowB1 = z + (long long)e1[edgeB] * D;  // x_j tile1

    const int koff = (lane < 16) ? 0 : 8;      // K offset of this lane's runs

    v8f acc[2][NTILES] = {};                   // two 16x128 hidden tiles, f32

    const v16h* Bv = (const v16h*)Bp;

#pragma unroll
    for (int s = 0; s < KSTEPS; ++s) {
        const int kg = s * 32 + koff;          // global K of first 8-run
        // k_steps 0-3 lie entirely in x_i, 4-7 entirely in x_j
        const float* src0 = (kg < D) ? (rowA0 + kg) : (rowB0 + (kg - D));
        const float* src1 = (kg < D) ? (rowA1 + kg) : (rowB1 + (kg - D));
        v16h a0 = make_a_frag(src0);
        v16h a1 = make_a_frag(src1);

        const v16h* bbase = Bv + (s * NTILES) * 32 + lane;
#pragma unroll
        for (int t = 0; t < NTILES; ++t) {
            v16h b = bbase[t * 32];            // one 32B load per lane, 2 WMMAs
            acc[0][t] = __builtin_amdgcn_wmma_f32_16x16x32_f16(
                false, a0, false, b, (short)0, acc[0][t], false, false);
            acc[1][t] = __builtin_amdgcn_wmma_f32_16x16x32_f16(
                false, a1, false, b, (short)0, acc[1][t], false, false);
        }
    }

    // ---- layer 2: out[M] = sigmoid( sum_N relu(h[M][N]+b1[N]) * W2[N] + b2 )
    // C/D layout: lanes 0-15 hold rows M=r (r=0..7) at N=lane; lanes 16-31
    // hold rows M=r+8 at N=lane-16.
    const int n0 = lane & 15;
    const float bias2 = b2[0];
    const int mbase = (lane < 16) ? 0 : 8;

#pragma unroll
    for (int p = 0; p < 2; ++p) {
        float sr[8] = {0.f, 0.f, 0.f, 0.f, 0.f, 0.f, 0.f, 0.f};
#pragma unroll
        for (int t = 0; t < NTILES; ++t) {
            const float bb  = b1[t * 16 + n0];
            const float w2v = W2[t * 16 + n0];
#pragma unroll
            for (int r = 0; r < 8; ++r) {
                float h = acc[p][t][r] + bb;
                h = h > 0.f ? h : 0.f;         // ReLU
                sr[r] += h * w2v;
            }
        }
        // butterfly-reduce across the 16 lanes of each half
#pragma unroll
        for (int r = 0; r < 8; ++r) {
            sr[r] += __shfl_xor(sr[r], 1, 16);
            sr[r] += __shfl_xor(sr[r], 2, 16);
            sr[r] += __shfl_xor(sr[r], 4, 16);
            sr[r] += __shfl_xor(sr[r], 8, 16);
        }
        if (n0 == 0) {
#pragma unroll
            for (int r = 0; r < 8; ++r) {
                float x = sr[r] + bias2;
                out[(tile0 + p) * 16 + mbase + r] = 1.0f / (1.0f + expf(-x));
            }
        }
    }
}

extern "C" void kernel_launch(void* const* d_in, const int* in_sizes, int n_in,
                              void* d_out, int out_size, void* d_ws, size_t ws_size,
                              hipStream_t stream) {
    const float* z  = (const float*)d_in[0];
    const int*   e  = (const int*)d_in[1];          // [2][E]
    const float* W1 = (const float*)d_in[2];        // [256][128]
    const float* b1 = (const float*)d_in[3];        // [128]
    const float* W2 = (const float*)d_in[4];        // [128][1]
    const float* b2 = (const float*)d_in[5];        // [1]
    float* out = (float*)d_out;

    _Float16* Bp = (_Float16*)d_ws;                 // 64 KB packed W1 fragments

    const int packN = KSTEPS * NTILES * 32 * 16;    // 32768
    pack_w1_kernel<<<(packN + 255) / 256, 256, 0, stream>>>(W1, Bp);

    const int nblocks = (PAIRS + WAVES_PB - 1) / WAVES_PB;  // 2344
    edge_mlp_kernel<<<nblocks, 256, 0, stream>>>(
        z, e, e + N_EDGES_, Bp, b1, W2, b2, out);
}
